// Model_57939108823230
// MI455X (gfx1250) — compile-verified, matrix operand order checked by reading, not run on hardware
//
#include <hip/hip_runtime.h>
#include <stdint.h>

// ---------------------------------------------------------------------------
// AFNO-gated patch-MoE forecaster for MI455X (gfx1250, wave32, WMMA).
// Heavy math done as bf16 WMMA (v_wmma_f32_16x16x32_bf16); fp32 weights are
// streamed once from HBM (619 MB -> ~27us floor @ 23.3 TB/s) and converted to
// bf16 in-register. Embeddings cached as bf16 in workspace (29.5 MB, L2-hot)
// and staged into LDS with GLOBAL_LOAD_ASYNC_TO_LDS_B128 (ASYNCcnt path),
// overlapped with the W-tile fp32->bf16 conversion.
// ---------------------------------------------------------------------------

#define B_SZ   128
#define SEQ    96
#define CIN    14
#define DM     512
#define NEXP   17
#define FLEN   48
#define HID    192
#define PREDC  1344   // 96*14
#define TOPK   5
#define LAM_   0.01f

typedef __attribute__((ext_vector_type(16))) __bf16 v16bf;
typedef __attribute__((ext_vector_type(8)))  float  v8f;

__constant__ int c_patch[NEXP]  = {2,3,4,5,6,7,8,9,10,12,13,16,19,24,32,48,96};
__constant__ int c_lp[NEXP]     = {48,32,24,20,16,14,12,11,10,8,8,6,6,4,3,2,1};
// prefix sums of Lp, unit = 128*512 = 65536 emb elements
__constant__ int c_embOff[NEXP] = {0,48,80,104,124,140,154,166,177,187,195,203,209,215,219,222,224};

struct ExpertPtrs {
  const float* ew[NEXP];
  const float* hw[NEXP];
  const float* hb[NEXP];
};

__device__ __forceinline__ uint16_t f2bf(float f) {
  uint32_t u = __builtin_bit_cast(uint32_t, f);
  u += 0x7FFFu + ((u >> 16) & 1u);          // round-to-nearest-even
  return (uint16_t)(u >> 16);
}

union Pack8 { uint32_t u[8]; v16bf v; };

// ---------------------------------------------------------------------------
__global__ void kZero(float* __restrict__ p, int n) {
  int i = blockIdx.x * blockDim.x + threadIdx.x;
  if (i < n) p[i] = 0.f;
}

// ---------------------------------------------------------------------------
// Gating stage 1: per (b,c): DFT (48 bins), complex MLP 48->192->17,
// softshrink+magnitude, atomic mean over channels into weights[B,17].
__global__ void kGate1(const float* __restrict__ x,
                       const float* __restrict__ w1, const float* __restrict__ b1,
                       const float* __restrict__ w2, const float* __restrict__ b2,
                       float* __restrict__ weights) {
  const int bc = blockIdx.x;
  const int b = bc / CIN, c = bc - b * CIN;
  __shared__ float xv[SEQ], xr[FLEN], xi[FLEN], o1r[HID], o1i[HID];
  const int t = threadIdx.x;

  if (t < SEQ) xv[t] = x[b * (SEQ * CIN) + t * CIN + c];
  __syncthreads();

  if (t < FLEN) {                       // rfft bins k=1..48 (DC dropped)
    int k = t + 1;
    float sr = 0.f, si = 0.f;
    for (int tt = 0; tt < SEQ; ++tt) {
      float ang = -6.28318530717958647f * (float)(k * tt) * (1.f / 96.f);
      float s, cs; __sincosf(ang, &s, &cs);
      sr += xv[tt] * cs; si += xv[tt] * s;
    }
    xr[t] = sr; xi[t] = si;
  }
  __syncthreads();

  {                                      // complex layer 1 + ReLU (t = 0..191)
    const float* w1r = w1; const float* w1i = w1 + FLEN * HID;
    float ar = b1[t], ai = b1[HID + t];
    for (int f = 0; f < FLEN; ++f) {
      float wr = w1r[f * HID + t], wi = w1i[f * HID + t];
      ar += xr[f] * wr - xi[f] * wi;
      ai += xi[f] * wr + xr[f] * wi;
    }
    o1r[t] = ar > 0.f ? ar : 0.f;
    o1i[t] = ai > 0.f ? ai : 0.f;
  }
  __syncthreads();

  if (t < NEXP) {                        // complex layer 2 + softshrink + |.|
    const float* w2r = w2; const float* w2i = w2 + HID * NEXP;
    float ar = b2[t], ai = b2[NEXP + t];
    for (int h = 0; h < HID; ++h) {
      float wr = w2r[h * NEXP + t], wi = w2i[h * NEXP + t];
      ar += o1r[h] * wr - o1i[h] * wi;
      ai += o1i[h] * wr + o1r[h] * wi;
    }
    float sr = fabsf(ar) - LAM_; sr = sr > 0.f ? sr : 0.f; sr = copysignf(sr, ar);
    float si = fabsf(ai) - LAM_; si = si > 0.f ? si : 0.f; si = copysignf(si, ai);
    unsafeAtomicAdd(&weights[b * NEXP + t], sqrtf(sr * sr + si * si) * (1.f / (float)CIN));
  }
}

// Gating stage 2: top-5 + softmax -> dense gates[B,17]
__global__ void kGate2(const float* __restrict__ weights, float* __restrict__ gates) {
  const int b = blockIdx.x;
  if (threadIdx.x != 0) return;
  float w[NEXP]; bool used[NEXP];
  for (int e = 0; e < NEXP; ++e) {
    w[e] = weights[b * NEXP + e]; used[e] = false; gates[b * NEXP + e] = 0.f;
  }
  float tv[TOPK]; int ti[TOPK];
  for (int s = 0; s < TOPK; ++s) {
    float best = -1e30f; int bi = 0;
    for (int e = 0; e < NEXP; ++e)
      if (!used[e] && w[e] > best) { best = w[e]; bi = e; }
    used[bi] = true; tv[s] = best; ti[s] = bi;
  }
  float mx = tv[0], den = 0.f, ex[TOPK];
  for (int s = 0; s < TOPK; ++s) { ex[s] = __expf(tv[s] - mx); den += ex[s]; }
  for (int s = 0; s < TOPK; ++s) gates[b * NEXP + ti[s]] = ex[s] / den;
}

// ---------------------------------------------------------------------------
// Embed GEMM per expert: emb[b,l,:] = xp[b,l,:] @ ew (+ PE), stored bf16.
// Block: 256 thr = 8 waves, tile = all 128 b-rows x 64 d-cols, fixed l.
__global__ void kEmbed(const float* __restrict__ x, ExpertPtrs P,
                       uint16_t* __restrict__ emb) {
  const int e  = blockIdx.z;
  const int l  = blockIdx.y;
  const int Lp = c_lp[e];
  if (l >= Lp) return;
  const int p  = c_patch[e];
  const int pC = p * CIN;
  const int K  = Lp * DM;
  const int nt = blockIdx.x * 64;
  const float* ew = P.ew[e];
  const size_t eOff = (size_t)c_embOff[e] * 65536u;

  __shared__ __align__(16) uint16_t aLds[128 * 32];
  __shared__ __align__(16) uint16_t wLds[64 * 32];

  const int t = threadIdx.x;
  const int wv = t >> 5, lane = t & 31, half = lane >> 4, rlo = lane & 15;

  v8f zero = {0.f,0.f,0.f,0.f,0.f,0.f,0.f,0.f};
  v8f accv[4] = {zero, zero, zero, zero};

  const int nK = (pC + 31) >> 5;
  for (int kb0 = 0; kb0 < nK; ++kb0) {
    const int kb = kb0 << 5;
    // A tile: rows b=0..127, cols j=kb..kb+31 of xp (replication-padded patchify)
    #pragma unroll
    for (int i = 0; i < 16; ++i) {
      int ei = t + (i << 8);
      int b = ei >> 5, jj = ei & 31, j = kb + jj;
      float v = 0.f;
      if (j < pC) {
        int pp = j / CIN;
        int c  = j - pp * CIN;
        int tt = l * p + pp; if (tt > SEQ - 1) tt = SEQ - 1;
        v = x[b * (SEQ * CIN) + tt * CIN + c];
      }
      aLds[ei] = f2bf(v);
    }
    // W tile transposed: wLds[d-nt][j-kb]; rows beyond pC are zero
    {
      int k = t >> 3, n0 = (t & 7) << 3, j = kb + k;
      float4 v0 = make_float4(0.f, 0.f, 0.f, 0.f), v1 = v0;
      if (j < pC) {
        const float4* src = (const float4*)(ew + (size_t)j * DM + nt + n0);
        v0 = src[0]; v1 = src[1];
      }
      wLds[(n0 + 0) * 32 + k] = f2bf(v0.x); wLds[(n0 + 1) * 32 + k] = f2bf(v0.y);
      wLds[(n0 + 2) * 32 + k] = f2bf(v0.z); wLds[(n0 + 3) * 32 + k] = f2bf(v0.w);
      wLds[(n0 + 4) * 32 + k] = f2bf(v1.x); wLds[(n0 + 5) * 32 + k] = f2bf(v1.y);
      wLds[(n0 + 6) * 32 + k] = f2bf(v1.z); wLds[(n0 + 7) * 32 + k] = f2bf(v1.w);
    }
    __syncthreads();

    Pack8 ap;
    #pragma unroll
    for (int r = 0; r < 8; ++r) {   // ISA A layout: k0 = 16*(r>=4) + 8*half + 2*(r&3)
      int k0 = ((r & 4) << 2) + (half << 3) + ((r & 3) << 1);
      ap.u[r] = *(const uint32_t*)&aLds[(wv * 16 + rlo) * 32 + k0];
    }
    #pragma unroll
    for (int j = 0; j < 4; ++j) {
      Pack8 bp;
      #pragma unroll
      for (int r = 0; r < 8; ++r) {
        int k0 = ((r & 4) << 2) + (half << 3) + ((r & 3) << 1);
        bp.u[r] = *(const uint32_t*)&wLds[(j * 16 + rlo) * 32 + k0];
      }
      accv[j] = __builtin_amdgcn_wmma_f32_16x16x32_bf16(
          false, ap.v, false, bp.v, (short)0, accv[j], false, false);
    }
    __syncthreads();
  }

  // Epilogue: add sinusoidal PE, store bf16 emb[b][l*512+d]
  #pragma unroll
  for (int j = 0; j < 4; ++j) {
    int d = nt + j * 16 + rlo;
    int kk = d & ~1;
    float freq = __expf((float)kk * (-9.21034037198f / 512.f));  // 10000^(-kk/512)
    float ang = (float)l * freq;
    float pe = (d & 1) ? __cosf(ang) : __sinf(ang);
    #pragma unroll
    for (int r = 0; r < 8; ++r) {     // C/D layout: m = r + 8*half, n = rlo
      int b = wv * 16 + r + (half << 3);
      emb[eOff + (size_t)b * K + (size_t)l * DM + d] = f2bf(accv[j][r] + pe);
    }
  }
}

// ---------------------------------------------------------------------------
// Head GEMM per expert: y = emb_flat[128,K] @ hw[K,1344] + hb;
// epilogue acc += gate*exp(y) via fp32 atomics. Each weight read exactly once.
// A tile staged via GLOBAL_LOAD_ASYNC_TO_LDS_B128 (ASYNCcnt), overlapped with
// the synchronous W-tile fp32->bf16 conversion through VGPRs.
__global__ void kHead(const uint16_t* __restrict__ emb, ExpertPtrs P,
                      const float* __restrict__ gates, float* __restrict__ acc) {
  const int e  = blockIdx.z;
  const int Lp = c_lp[e];
  const int K  = Lp * DM;
  const int nt = blockIdx.x * 64;
  const float* hw = P.hw[e];
  const float* hb = P.hb[e];
  // 64-bit uniform base of this expert's emb block (bf16 elements)
  const uint64_t embBase = (uint64_t)(uintptr_t)(emb + (size_t)c_embOff[e] * 65536u);

  __shared__ __align__(16) uint16_t aLds[128 * 32];
  __shared__ __align__(16) uint16_t wLds[64 * 32];

  const int t = threadIdx.x;
  const int wv = t >> 5, lane = t & 31, half = lane >> 4, rlo = lane & 15;

  // LDS byte addresses (addrspace(3) offset) for this thread's two 16B chunks
  const uint32_t ldsABase = (uint32_t)(uintptr_t)(&aLds[0]);

  v8f zero = {0.f,0.f,0.f,0.f,0.f,0.f,0.f,0.f};
  v8f accv[4] = {zero, zero, zero, zero};

  const int nK = K >> 5;
  for (int kb0 = 0; kb0 < nK; ++kb0) {
    const int kb = kb0 << 5;
    // ---- A tile: 8KB, 512 x b128 async copies memory->LDS (no VGPR transit)
    #pragma unroll
    for (int i = 0; i < 2; ++i) {
      int c = t + (i << 8);                       // b128 chunk id, 0..511
      // chunk -> (row b = c>>2, 16B sub-chunk (c&3)); byte offsets:
      uint32_t gOff = (uint32_t)((c >> 2) * (K << 1) + (kb << 1) + ((c & 3) << 4));
      uint32_t lOff = ldsABase + ((uint32_t)c << 4);
      asm volatile("global_load_async_to_lds_b128 %0, %1, %2"
                   :: "v"(lOff), "v"(gOff), "s"(embBase) : "memory");
    }
    // ---- W tile (overlaps the async copy): fp32 -> bf16, transposed wLds[n][k]
    {
      int k = t >> 3, n0 = (t & 7) << 3;
      const float4* src = (const float4*)(hw + (size_t)(kb + k) * PREDC + nt + n0);
      float4 v0 = src[0], v1 = src[1];
      if (kb + 32 < K)   // speculative prefetch of next W tile row
        __builtin_prefetch(hw + (size_t)(kb + 32 + k) * PREDC + nt + n0, 0, 0);
      wLds[(n0 + 0) * 32 + k] = f2bf(v0.x); wLds[(n0 + 1) * 32 + k] = f2bf(v0.y);
      wLds[(n0 + 2) * 32 + k] = f2bf(v0.z); wLds[(n0 + 3) * 32 + k] = f2bf(v0.w);
      wLds[(n0 + 4) * 32 + k] = f2bf(v1.x); wLds[(n0 + 5) * 32 + k] = f2bf(v1.y);
      wLds[(n0 + 6) * 32 + k] = f2bf(v1.z); wLds[(n0 + 7) * 32 + k] = f2bf(v1.w);
    }
    asm volatile("s_wait_asynccnt 0" ::: "memory");
    __syncthreads();

    Pack8 ap;
    #pragma unroll
    for (int r = 0; r < 8; ++r) {
      int k0 = ((r & 4) << 2) + (half << 3) + ((r & 3) << 1);
      ap.u[r] = *(const uint32_t*)&aLds[(wv * 16 + rlo) * 32 + k0];
    }
    #pragma unroll
    for (int j = 0; j < 4; ++j) {
      Pack8 bp;
      #pragma unroll
      for (int r = 0; r < 8; ++r) {
        int k0 = ((r & 4) << 2) + (half << 3) + ((r & 3) << 1);
        bp.u[r] = *(const uint32_t*)&wLds[(j * 16 + rlo) * 32 + k0];
      }
      accv[j] = __builtin_amdgcn_wmma_f32_16x16x32_bf16(
          false, ap.v, false, bp.v, (short)0, accv[j], false, false);
    }
    __syncthreads();
  }

  // Epilogue: acc[b,n] += gates[b,e] * exp(y + hb[n])
  float g[8];
  #pragma unroll
  for (int r = 0; r < 8; ++r)
    g[r] = gates[(wv * 16 + r + (half << 3)) * NEXP + e];
  #pragma unroll
  for (int j = 0; j < 4; ++j) {
    int n = nt + j * 16 + rlo;
    float hbn = hb[n];
    #pragma unroll
    for (int r = 0; r < 8; ++r) {
      int b = wv * 16 + r + (half << 3);
      unsafeAtomicAdd(&acc[(size_t)b * PREDC + n], g[r] * __expf(accv[j][r] + hbn));
    }
  }
}

// ---------------------------------------------------------------------------
__global__ void kFinal(const float* __restrict__ acc, float* __restrict__ out, int n) {
  int i = blockIdx.x * blockDim.x + threadIdx.x;
  if (i >= n) return;
  float a = acc[i];
  out[i] = logf(a == 0.f ? 2.2204460492503131e-16f : a);
}

// ---------------------------------------------------------------------------
extern "C" void kernel_launch(void* const* d_in, const int* in_sizes, int n_in,
                              void* d_out, int out_size, void* d_ws, size_t ws_size,
                              hipStream_t stream) {
  const float* x  = (const float*)d_in[0];
  const float* w1 = (const float*)d_in[1];
  const float* b1 = (const float*)d_in[2];
  const float* w2 = (const float*)d_in[3];
  const float* b2 = (const float*)d_in[4];
  ExpertPtrs P;
  for (int i = 0; i < NEXP; ++i) {
    P.ew[i] = (const float*)d_in[5 + i];
    P.hw[i] = (const float*)d_in[5 + NEXP + i];
    P.hb[i] = (const float*)d_in[5 + 2 * NEXP + i];
  }

  // Workspace layout (floats): [weights 2176][gates 2176][acc 172032][emb bf16 14.75M]
  float*    wsF      = (float*)d_ws;
  float*    wWeights = wsF;
  float*    wGates   = wsF + 2176;
  float*    wAcc     = wsF + 4352;
  uint16_t* wEmb     = (uint16_t*)(wsF + 176384);

  kZero <<<(176384 + 255) / 256, 256, 0, stream>>>(wsF, 176384);
  kGate1<<<B_SZ * CIN, HID, 0, stream>>>(x, w1, b1, w2, b2, wWeights);
  kGate2<<<B_SZ, 32, 0, stream>>>(wWeights, wGates);
  kEmbed<<<dim3(8, 48, NEXP), 256, 0, stream>>>(x, P, wEmb);
  kHead <<<dim3(21, 1, NEXP), 256, 0, stream>>>(wEmb, P, wGates, wAcc);
  kFinal<<<(B_SZ * PREDC + 255) / 256, 256, 0, stream>>>(wAcc, (float*)d_out, B_SZ * PREDC);
}
